// KalmanNet_15977278341352
// MI455X (gfx1250) — matrix operand order repaired and checked by reading, not compiled, not run on hardware
//
#include <hip/hip_runtime.h>

typedef __attribute__((ext_vector_type(16))) __bf16 v16bf;
typedef __attribute__((ext_vector_type(8)))  float  v8f;
typedef unsigned short u16;

// Problem dims
constexpr int Bx   = 128;
constexpr int Tx   = 512;
constexpr int INx  = 128;
constexpr int Hx   = 256;
constexpr int G4H  = 1024;   // 4*H
constexpr int KGN  = 512;    // OUT*IN

// Workspace layout (bytes)
constexpr size_t OFF_XA  = 0;                          // x A-frags bf16: 4096*4*512
constexpr size_t OFF_WIH = OFF_XA  + 8388608ull * 2;   // W_ih B-frags: 64*4*512
constexpr size_t OFF_WHH = OFF_WIH + 131072ull  * 2;   // W_hh B-frags: 64*8*512
constexpr size_t OFF_WFC = OFF_WHH + 262144ull  * 2;   // W_fc B-frags: 32*8*512
constexpr size_t OFF_XG  = OFF_WFC + 131072ull  * 2;   // xg fp32 [T][B][4H]
constexpr size_t OFF_HSA = OFF_XG  + 67108864ull * 4;  // hs A-frags bf16: 4096*8*512
constexpr size_t OFF_KG  = OFF_HSA + 16777216ull * 2;  // kg fp32 [T][B][512]
constexpr size_t OFF_BAR = OFF_KG  + 33554432ull * 4;  // 8 groups x {arrive, generation}

// ---------- helpers ----------
__device__ __forceinline__ u16 f2bf(float f) {
  union { float f; unsigned u; } v; v.f = f;
  unsigned u = v.u;
  u += 0x7FFFu + ((u >> 16) & 1u);   // round-to-nearest-even
  return (u16)(u >> 16);
}

// 16-bit A-matrix 16x32 fragment layout (ISA 7.12.2)
__device__ __forceinline__ int a_frag_k(int lane, int j) {
  int r = j >> 1, lo = j & 1;
  int base = (r < 4) ? (2 * r) : (16 + 2 * (r - 4));
  return base + ((lane >> 4) << 3) + lo;
}
__device__ __forceinline__ void a_frag_pos(int kin, int& hi16, int& j) {
  int hi8 = (kin >> 3) & 1;
  hi16 = hi8;
  int rem = kin - (hi8 << 3);
  int r = (rem < 8) ? (rem >> 1) : (4 + ((rem - 16) >> 1));
  j = (r << 1) | (kin & 1);
}

__device__ __forceinline__ float sig_(float x) { return 1.0f / (1.0f + __expf(-x)); }
__device__ __forceinline__ float tanh_(float x) {
  float ax = fabsf(x);
  float e  = __expf(-2.0f * ax);
  float t  = (1.0f - e) / (1.0f + e);
  return copysignf(t, x);
}

// ---------- pack kernels ----------
__global__ __launch_bounds__(256) void pack_x_kernel(const float* __restrict__ x,
                                                     u16* __restrict__ xA) {
  size_t e = (size_t)blockIdx.x * 256 + threadIdx.x;   // < 8,388,608
  int j    = (int)(e & 15);
  int lane = (int)((e >> 4) & 31);
  size_t fi = e >> 9;
  int kt = (int)(fi & 3);
  int mt = (int)(fi >> 2);
  int t = mt >> 3, bblk = mt & 7;
  int b = bblk * 16 + (lane & 15);
  int k = kt * 32 + a_frag_k(lane, j);
  xA[e] = f2bf(x[((size_t)b * Tx + t) * INx + k]);
}

// W [N][K] fp32 -> B-frags bf16 (B = W^T, 32x16 tiles)
__global__ __launch_bounds__(256) void pack_w_kernel(const float* __restrict__ W,
                                                     u16* __restrict__ dst,
                                                     int KT, int K) {
  size_t e = (size_t)blockIdx.x * 256 + threadIdx.x;
  int j    = (int)(e & 15);
  int lane = (int)((e >> 4) & 31);
  size_t fi = e >> 9;
  int kt = (int)(fi % (size_t)KT);
  int nt = (int)(fi / (size_t)KT);
  int n = nt * 16 + (lane & 15);
  int k = kt * 32 + a_frag_k(lane, j);
  dst[e] = f2bf(W[(size_t)n * K + k]);
}

__global__ void init_bar_kernel(unsigned* __restrict__ barv) {
  if (threadIdx.x < 16) barv[threadIdx.x] = 0u;
}

// ---------- xg = x @ W_ih^T + b_ih  (M=65536, N=1024, K=128) ----------
__global__ __launch_bounds__(256) void xgemm_kernel(const u16* __restrict__ xA,
                                                    const u16* __restrict__ WB,
                                                    const float* __restrict__ bias,
                                                    float* __restrict__ xg) {
  int gw   = blockIdx.x * 8 + (threadIdx.x >> 5);   // 65536 waves
  int lane = threadIdx.x & 31;
  int mt = gw >> 4, ng = gw & 15;
  v8f acc[4];
#pragma unroll
  for (int q = 0; q < 4; q++) {
    float bv = bias[(ng * 4 + q) * 16 + (lane & 15)];
#pragma unroll
    for (int r = 0; r < 8; r++) acc[q][r] = bv;
  }
#pragma unroll
  for (int kt = 0; kt < 4; kt++) {
    v16bf a = *(const v16bf*)(xA + ((size_t)(mt * 4 + kt) * 512) + lane * 16);
#pragma unroll
    for (int q = 0; q < 4; q++) {
      int nt = ng * 4 + q;
      v16bf b = *(const v16bf*)(WB + ((size_t)(nt * 4 + kt) * 512) + lane * 16);
      acc[q] = __builtin_amdgcn_wmma_f32_16x16x32_bf16(false, a, false, b, (short)0,
                                                       acc[q], false, false);
    }
  }
  int t = mt >> 3, b0 = (mt & 7) * 16, hi = lane >> 4;
#pragma unroll
  for (int q = 0; q < 4; q++) {
    int col = (ng * 4 + q) * 16 + (lane & 15);
#pragma unroll
    for (int r = 0; r < 8; r++)
      xg[((size_t)t * Bx + b0 + r + 8 * hi) * G4H + col] = acc[q][r];
  }
}

// ---------- LSTM scan, cluster-cooperative, W_hh slice resident in LDS ----------
// 32 WGs = 8 batch tiles x 4 column groups. Each WG: 64 hidden cols, all 4 gates.
// The WG's 128 KB W_hh B-fragment slice is step-invariant -> staged to LDS once;
// per-step WMMA B operands come from ds_load_b128 instead of L2.
// h exchanged between the 4 WGs of a batch tile through L2 via hsA[t] fragments;
// per-step sync = global sense-reversing barrier + s_cluster_barrier.
__global__ __launch_bounds__(256) void lstm_kernel(const float* __restrict__ xg,
                                                   const u16* __restrict__ WB,
                                                   const float* __restrict__ b_hh,
                                                   u16* __restrict__ hsA,
                                                   unsigned* __restrict__ barv) {
  __shared__ __align__(16) u16 wlds[16][8][512];   // [local nt][kt][frag] = 128 KB
  __shared__ float g_lds[16][4][64];               // [row][gate][local col] = 16 KB

  int tid = threadIdx.x, lane = tid & 31, w = tid >> 5;
  int bblk = blockIdx.x >> 2, cg = blockIdx.x & 3;
  int b0 = bblk * 16;
  int hi = lane >> 4;
  int ct_l  = w >> 1;          // local column tile 0..3
  int gbase = (w & 1) * 2;     // this wave computes gates {gbase, gbase+1}

  // ---- stage this WG's W_hh B-fragments into LDS (16-byte copies) ----
  {
    uint4* dst = (uint4*)wlds;
    for (int f = tid; f < 8192; f += 256) {
      int ln  = f >> 9;          // local nt: gate*4 + ct_l
      int rem = f & 511;
      int kt  = rem >> 6;
      int e8  = rem & 63;
      int nt  = ((ln >> 2) * 16) + cg * 4 + (ln & 3);
      dst[f] = *((const uint4*)(WB + ((size_t)(nt * 8 + kt) * 512)) + e8);
    }
  }

  int   ntq[2], lnq[2];
  float bh[2];
#pragma unroll
  for (int q = 0; q < 2; q++) {
    int gate = gbase + q;
    ntq[q] = gate * 16 + cg * 4 + ct_l;
    lnq[q] = gate * 4 + ct_l;
    bh[q]  = b_hh[ntq[q] * 16 + (lane & 15)];
  }

  // gate-combine mapping: thread owns (col_l, rows m0..m0+3)
  int col_l = tid & 63;
  int m0    = (tid >> 6) * 4;
  int hcol  = cg * 64 + col_l;
  int ktw   = hcol >> 5;
  int hi16, jj;
  a_frag_pos(hcol & 31, hi16, jj);
  float c[4] = {0.f, 0.f, 0.f, 0.f};

  unsigned* bar = barv + (size_t)bblk * 2;
  unsigned* gen = bar + 1;

  __syncthreads();   // W_hh slice resident

  for (int t = 0; t < Tx; t++) {
    v8f acc[2];
    // ---- acc init from precomputed input projection ----
#pragma unroll
    for (int q = 0; q < 2; q++) {
      int col = ntq[q] * 16 + (lane & 15);
      const float* xgp = xg + ((size_t)t * Bx + b0 + 8 * hi) * G4H + col;
#pragma unroll
      for (int r = 0; r < 8; r++) acc[q][r] = xgp[(size_t)r * G4H] + bh[q];
      if (t + 1 < Tx) __builtin_prefetch(xgp + (size_t)Bx * G4H, 0, 1);
    }
    // ---- + h_{t-1} @ W_hh^T : A frags from L2 (peer WGs), B frags from LDS ----
    if (t > 0) {
      const u16* hb = hsA + ((size_t)((t - 1) * 8 + bblk) * 8) * 512;
#pragma unroll
      for (int kt = 0; kt < 8; kt++) {
        v16bf a = *(const v16bf*)(hb + (size_t)kt * 512 + lane * 16);
#pragma unroll
        for (int q = 0; q < 2; q++) {
          v16bf bf = *(const v16bf*)(&wlds[lnq[q]][kt][lane * 16]);
          acc[q] = __builtin_amdgcn_wmma_f32_16x16x32_bf16(false, a, false, bf,
                                                           (short)0, acc[q],
                                                           false, false);
        }
      }
    }
    // ---- stage gate pre-activations to LDS ----
#pragma unroll
    for (int q = 0; q < 2; q++) {
      int cl = ct_l * 16 + (lane & 15);
#pragma unroll
      for (int r = 0; r < 8; r++) g_lds[r + 8 * hi][gbase + q][cl] = acc[q][r];
    }
    __syncthreads();

    // ---- combine gates, update c, emit h fragments ----
#pragma unroll
    for (int mm = 0; mm < 4; mm++) {
      int m = m0 + mm;
      float iv = sig_(g_lds[m][0][col_l]);
      float fv = sig_(g_lds[m][1][col_l]);
      float gv = tanh_(g_lds[m][2][col_l]);
      float ov = sig_(g_lds[m][3][col_l]);
      c[mm] = fv * c[mm] + iv * gv;
      float hv = ov * tanh_(c[mm]);
      int lane2 = m + (hi16 << 4);
      hsA[(((size_t)(t * 8 + bblk) * 8 + ktw) * 512) + lane2 * 16 + jj] = f2bf(hv);
    }
    __syncthreads();   // g_lds reusable next step

    // ---- sync the 4 WGs of this batch tile ----
    if (tid == 0) {
      __threadfence();                               // release h stores
      unsigned g   = *(volatile unsigned*)gen;
      unsigned arr = atomicAdd(bar, 1u) + 1u;
      if (arr == 4u) {
        *(volatile unsigned*)bar = 0u;
        __threadfence();
        atomicAdd(gen, 1u);
      } else {
        while (*(volatile unsigned*)gen == g) __builtin_amdgcn_s_sleep(1);
      }
    }
    __syncthreads();
    __builtin_amdgcn_s_cluster_barrier();  // s_barrier_signal/wait -3 (NOP off-cluster)
    __threadfence();                       // acquire peers' h stores
  }
}

// ---------- kg = hs @ W_fc^T + b_fc  (M=65536, N=512, K=256) ----------
__global__ __launch_bounds__(256) void fcgemm_kernel(const u16* __restrict__ hsA,
                                                     const u16* __restrict__ WB,
                                                     const float* __restrict__ bias,
                                                     float* __restrict__ kg) {
  int gw   = blockIdx.x * 8 + (threadIdx.x >> 5);   // 32768 waves
  int lane = threadIdx.x & 31;
  int mt = gw >> 3, ng = gw & 7;
  v8f acc[4];
#pragma unroll
  for (int q = 0; q < 4; q++) {
    float bv = bias[(ng * 4 + q) * 16 + (lane & 15)];
#pragma unroll
    for (int r = 0; r < 8; r++) acc[q][r] = bv;
  }
#pragma unroll
  for (int kt = 0; kt < 8; kt++) {
    v16bf a = *(const v16bf*)(hsA + ((size_t)(mt * 8 + kt) * 512) + lane * 16);
#pragma unroll
    for (int q = 0; q < 4; q++) {
      int nt = ng * 4 + q;
      v16bf b = *(const v16bf*)(WB + ((size_t)(nt * 8 + kt) * 512) + lane * 16);
      acc[q] = __builtin_amdgcn_wmma_f32_16x16x32_bf16(false, a, false, b, (short)0,
                                                       acc[q], false, false);
    }
  }
  int t = mt >> 3, b0 = (mt & 7) * 16, hi = lane >> 4;
#pragma unroll
  for (int q = 0; q < 4; q++) {
    int col = (ng * 4 + q) * 16 + (lane & 15);
#pragma unroll
    for (int r = 0; r < 8; r++)
      kg[((size_t)t * Bx + b0 + r + 8 * hi) * KGN + col] = acc[q][r];
  }
}

// ---------- Kalman scan: one wave per batch element ----------
__global__ __launch_bounds__(32) void kalman_kernel(const float* __restrict__ x,
                                                    const float* __restrict__ kg,
                                                    const float* __restrict__ dA,
                                                    const float* __restrict__ dC,
                                                    float* __restrict__ out) {
  int b = blockIdx.x, lane = threadIdx.x;
  float Cr[4][4];
#pragma unroll
  for (int ii = 0; ii < 4; ii++) {
    int i = lane * 4 + ii;
#pragma unroll
    for (int o = 0; o < 4; o++) Cr[ii][o] = dC[i * 4 + o];
  }
  float Am[16];
#pragma unroll
  for (int e = 0; e < 16; e++) Am[e] = dA[e];

  float s[4] = {0.f, 0.f, 0.f, 0.f};
  for (int t = 0; t < Tx; t++) {
    float prev[4];
#pragma unroll
    for (int o = 0; o < 4; o++)
      prev[o] = s[0] * Am[0 * 4 + o] + s[1] * Am[1 * 4 + o] +
                s[2] * Am[2 * 4 + o] + s[3] * Am[3 * 4 + o];

    const float* xp = x + ((size_t)b * Tx + t) * INx + lane * 4;
    float innov[4];
#pragma unroll
    for (int ii = 0; ii < 4; ii++)
      innov[ii] = xp[ii] - (prev[0] * Cr[ii][0] + prev[1] * Cr[ii][1] +
                            prev[2] * Cr[ii][2] + prev[3] * Cr[ii][3]);

    const float* kp = kg + ((size_t)t * Bx + b) * KGN;
#pragma unroll
    for (int o = 0; o < 4; o++) {
      const float* kpo = kp + o * INx + lane * 4;
      float v = kpo[0] * innov[0] + kpo[1] * innov[1] +
                kpo[2] * innov[2] + kpo[3] * innov[3];
#pragma unroll
      for (int off = 16; off >= 1; off >>= 1) v += __shfl_xor(v, off, 32);
      s[o] = prev[o] + v;
    }
    if (lane == 0) {
      float4 ov; ov.x = s[0]; ov.y = s[1]; ov.z = s[2]; ov.w = s[3];
      *(float4*)(out + ((size_t)b * Tx + t) * 4) = ov;
    }
  }
}

extern "C" void kernel_launch(void* const* d_in, const int* in_sizes, int n_in,
                              void* d_out, int out_size, void* d_ws, size_t ws_size,
                              hipStream_t stream) {
  (void)in_sizes; (void)n_in; (void)out_size; (void)ws_size;
  const float* x    = (const float*)d_in[0];
  const float* W_ih = (const float*)d_in[1];
  const float* W_hh = (const float*)d_in[2];
  const float* b_ih = (const float*)d_in[3];
  const float* b_hh = (const float*)d_in[4];
  const float* W_fc = (const float*)d_in[5];
  const float* b_fc = (const float*)d_in[6];
  const float* Amat = (const float*)d_in[7];
  const float* Cmat = (const float*)d_in[8];

  char* ws = (char*)d_ws;
  u16*      xA    = (u16*)     (ws + OFF_XA);
  u16*      WihB  = (u16*)     (ws + OFF_WIH);
  u16*      WhhB  = (u16*)     (ws + OFF_WHH);
  u16*      WfcB  = (u16*)     (ws + OFF_WFC);
  float*    xgbuf = (float*)   (ws + OFF_XG);
  u16*      hsA   = (u16*)     (ws + OFF_HSA);
  float*    kgbuf = (float*)   (ws + OFF_KG);
  unsigned* barv  = (unsigned*)(ws + OFF_BAR);

  // pack operands to fragment-major bf16
  pack_x_kernel<<<32768, 256, 0, stream>>>(x, xA);
  pack_w_kernel<<<512,  256, 0, stream>>>(W_ih, WihB, 4, INx);
  pack_w_kernel<<<1024, 256, 0, stream>>>(W_hh, WhhB, 8, Hx);
  pack_w_kernel<<<512,  256, 0, stream>>>(W_fc, WfcB, 8, Hx);
  init_bar_kernel<<<1, 32, 0, stream>>>(barv);

  // xg = x @ W_ih^T + b_ih (time-major [T][B][4H])
  xgemm_kernel<<<8192, 256, 0, stream>>>(xA, WihB, b_ih, xgbuf);

  // cluster-cooperative LSTM scan -> hs fragments
  lstm_kernel<<<32, 256, 0, stream>>>(xgbuf, WhhB, b_hh, hsA, barv);

  // kg = hs @ W_fc^T + b_fc (time-major [T][B][512])
  fcgemm_kernel<<<4096, 256, 0, stream>>>(hsA, WfcB, b_fc, kgbuf);

  // Kalman recurrence -> pred [B,T,OUT]
  kalman_kernel<<<Bx, 32, 0, stream>>>(x, kgbuf, Amat, Cmat, (float*)d_out);
}
